// PointCloudTCN_11261404250709
// MI455X (gfx1250) — matrix operand order repaired and checked by reading, not compiled, not run on hardware
//
#include <hip/hip_runtime.h>
#include <hip/hip_bf16.h>

typedef __attribute__((ext_vector_type(16))) _Float16 v16h;
typedef __attribute__((ext_vector_type(8)))  float    v8f;

// ---------------------------------------------------------------------------
// WMMA fragment loaders (wave32; A/B in contiguous k order chosen here, C/D per ISA layout)
// A: 16x32 f16.  lane L holds M = L&15.  VGPR p: K = kh+2p, kh=(L<16?0:16)
//                (contiguous, matches B).
// B: 32x16 f16 (K x N). lane L holds N = L&15, K = (L<16?0:16)+e, e=0..15
//    contiguous -> store weights transposed [Nout][K] so reads are contiguous.
// C/D: f32, lane L col N=L&15, VGPR v row M = v + (L<16?0:8).
// ---------------------------------------------------------------------------
__device__ __forceinline__ v16h frag_a(const _Float16* A, int ldA, int lane, int kofs) {
  const int m  = lane & 15;
  const int kh = (lane & 16) ? 16 : 0;
  const _Float16* row = A + m * ldA + kofs + kh;
  v16h a;
#pragma unroll
  for (int p = 0; p < 8; ++p) {
    const int K = 2 * p;
    a[2 * p]     = row[K];
    a[2 * p + 1] = row[K + 1];
  }
  return a;
}

__device__ __forceinline__ v16h frag_b(const _Float16* Wt, int ldW, int lane, int t, int k) {
  const int n  = (lane & 15) + t * 16;
  const int kb = k * 32 + ((lane & 16) ? 16 : 0);
  const _Float16* row = Wt + n * ldW + kb;
  v16h b;
#pragma unroll
  for (int e = 0; e < 16; ++e) b[e] = row[e];
  return b;
}

template <int KT, int NT>
__device__ __forceinline__ void wmma_gemm(const _Float16* A, int ldA,
                                          const _Float16* Wt, int ldW,
                                          const float* bias, int lane, v8f* acc) {
  const int ncol = lane & 15;
#pragma unroll
  for (int t = 0; t < NT; ++t) {
    const float bv = bias[t * 16 + ncol];
    v8f c;
#pragma unroll
    for (int v = 0; v < 8; ++v) c[v] = bv;
#pragma unroll
    for (int k = 0; k < KT; ++k) {
      v16h a = frag_a(A, ldA, lane, k * 32);
      v16h b = frag_b(Wt, ldW, lane, t, k);
      c = __builtin_amdgcn_wmma_f32_16x16x32_f16(false, a, false, b, (short)0, c,
                                                 false, false);
    }
    acc[t] = c;
  }
}

template <int NT>
__device__ __forceinline__ void store_relu(_Float16* An, int ldA, int lane, const v8f* acc) {
  const int ncol = lane & 15;
  const int mb   = (lane & 16) ? 8 : 0;
#pragma unroll
  for (int t = 0; t < NT; ++t)
#pragma unroll
    for (int v = 0; v < 8; ++v) {
      float x = acc[t][v];
      An[(mb + v) * ldA + t * 16 + ncol] = (_Float16)(x > 0.f ? x : 0.f);
    }
}

// ---------------------------------------------------------------------------
// Edge-side 3-layer MLP (rel MLP / edge-weight MLP) with gather + scatter-add.
// NH: # of node features gathered (14 = h[dst]|h[src], 0 = none)
// DINE: edge-feature width; HID: true hidden dim; HP: padded hidden (mult 32)
// EOUT: true out dim (<=8); MODE: feature assembly; SIG: sigmoid; DOAGG: agg.
// ---------------------------------------------------------------------------
template <int NH, int DINE, int HID, int HP, int EOUT, int MODE, bool SIG, bool DOAGG>
__global__ __launch_bounds__(128) void edge_mlp_kernel(
    const float* __restrict__ hin, const int* __restrict__ srcI,
    const int* __restrict__ dstI, const float* __restrict__ f0,
    const float* __restrict__ f1, const float* __restrict__ f2,
    const float* __restrict__ f3, const float* __restrict__ f4,
    const float* __restrict__ W1, const float* __restrict__ B1,
    const float* __restrict__ W2, const float* __restrict__ B2,
    const float* __restrict__ W3, const float* __restrict__ B3,
    float* __restrict__ eout, float* __restrict__ agg, int E) {
  constexpr int NT  = HP / 16;
  constexpr int KT2 = HP / 32;
  constexpr int DIN = NH + DINE;

  __shared__ _Float16 sW1t[HP][32];
  __shared__ _Float16 sW2t[HP][HP];
  __shared__ _Float16 sW3t[16][HP];
  __shared__ float    sB1[HP], sB2[HP], sB3[16];
  __shared__ _Float16 sA0[4][16][32];
  __shared__ _Float16 sA1[4][16][HP];
  __shared__ _Float16 sA2[4][16][HP];

  const int tid = threadIdx.x, w = tid >> 5, lane = tid & 31;

  for (int i = tid; i < HP * 32; i += 128) {
    int n = i >> 5, k = i & 31;
    sW1t[n][k] = (_Float16)((k < DIN && n < HID) ? W1[k * HID + n] : 0.f);
  }
  for (int i = tid; i < HP * HP; i += 128) {
    int n = i / HP, k = i % HP;
    sW2t[n][k] = (_Float16)((k < HID && n < HID) ? W2[k * HID + n] : 0.f);
  }
  for (int i = tid; i < 16 * HP; i += 128) {
    int n = i / HP, k = i % HP;
    sW3t[n][k] = (_Float16)((k < HID && n < EOUT) ? W3[k * EOUT + n] : 0.f);
  }
  for (int i = tid; i < HP; i += 128) {
    sB1[i] = (i < HID) ? B1[i] : 0.f;
    sB2[i] = (i < HID) ? B2[i] : 0.f;
  }
  if (tid < 16) sB3[tid] = (tid < EOUT) ? B3[tid] : 0.f;
  __syncthreads();

  const int ntiles = (E + 15) >> 4;
  const int stride = gridDim.x * 4;
  const int iters  = (ntiles + stride - 1) / stride;

  for (int it = 0; it < iters; ++it) {
    const int tile = (it * gridDim.x + blockIdx.x) * 4 + w;  // wave-uniform
    if (tile >= ntiles) continue;
    const int base = tile << 4;

    // ---- gather 16 rows x 32 cols into sA0[w] (f32 -> f16) ----
    for (int idx = lane; idx < 16 * 32; idx += 32) {
      const int r = idx >> 5, c = idx & 31;
      const int e = base + r;
      float v = 0.f;
      if (e < E && c < DIN) {
        if (NH > 0 && c < 7) {
          v = hin[dstI[e] * 7 + c];
        } else if (NH > 0 && c < 14) {
          v = hin[srcI[e] * 7 + (c - 7)];
        } else {
          const int fc = c - NH;
          if (MODE == 0) {
            v = f0[(size_t)e * DINE + fc];
          } else if (MODE == 1) {  // [ew, edge_attr, e1, e2, e3] = 17
            if (fc == 0)        v = f4[e];
            else if (fc < 5)    v = f0[(size_t)e * 4 + fc - 1];
            else if (fc < 9)    v = f1[(size_t)e * 4 + fc - 5];
            else if (fc < 13)   v = f2[(size_t)e * 4 + fc - 9];
            else                v = f3[(size_t)e * 4 + fc - 13];
          } else {                 // MODE 2: e_init = [edge_attr, e1, e2, e3] = 16
            if (fc < 4)         v = f0[(size_t)e * 4 + fc];
            else if (fc < 8)    v = f1[(size_t)e * 4 + fc - 4];
            else if (fc < 12)   v = f2[(size_t)e * 4 + fc - 8];
            else                v = f3[(size_t)e * 4 + fc - 12];
          }
        }
      }
      sA0[w][r][c] = (_Float16)v;
    }
    __builtin_amdgcn_wave_barrier();

    v8f acc[NT];
    wmma_gemm<1, NT>(&sA0[w][0][0], 32, &sW1t[0][0], 32, sB1, lane, acc);
    __builtin_amdgcn_wave_barrier();
    store_relu<NT>(&sA1[w][0][0], HP, lane, acc);
    __builtin_amdgcn_wave_barrier();
    wmma_gemm<KT2, NT>(&sA1[w][0][0], HP, &sW2t[0][0], HP, sB2, lane, acc);
    __builtin_amdgcn_wave_barrier();
    store_relu<NT>(&sA2[w][0][0], HP, lane, acc);
    __builtin_amdgcn_wave_barrier();
    v8f o[1];
    wmma_gemm<KT2, 1>(&sA2[w][0][0], HP, &sW3t[0][0], HP, sB3, lane, o);

    // ---- write e_new and scatter-add into agg[dst] ----
    const int n  = lane & 15;
    const int mb = (lane & 16) ? 8 : 0;
    if (n < EOUT) {
#pragma unroll
      for (int v = 0; v < 8; ++v) {
        const int e = base + mb + v;
        if (e < E) {
          float x = o[0][v];
          if (SIG) x = 1.f / (1.f + expf(-x));
          eout[(size_t)e * EOUT + n] = x;
          if (DOAGG) atomicAdd(&agg[(size_t)dstI[e] * EOUT + n], x);
        }
      }
    }
  }
}

// ---------------------------------------------------------------------------
// Node-side 3-layer MLP: in = [hin(7) | agg(EIN)], out = mlp (+res) (sigmoid?)
// ---------------------------------------------------------------------------
template <int EIN, int HID, int HP, int EOUT, bool SIG, bool RES>
__global__ __launch_bounds__(128) void node_mlp_kernel(
    const float* __restrict__ hin, const float* __restrict__ agg,
    const float* __restrict__ W1, const float* __restrict__ B1,
    const float* __restrict__ W2, const float* __restrict__ B2,
    const float* __restrict__ W3, const float* __restrict__ B3,
    const float* __restrict__ res, float* __restrict__ out, int N) {
  constexpr int NT  = HP / 16;
  constexpr int KT2 = HP / 32;
  constexpr int DIN = 7 + EIN;

  __shared__ _Float16 sW1t[HP][32];
  __shared__ _Float16 sW2t[HP][HP];
  __shared__ _Float16 sW3t[16][HP];
  __shared__ float    sB1[HP], sB2[HP], sB3[16];
  __shared__ _Float16 sA0[4][16][32];
  __shared__ _Float16 sA1[4][16][HP];
  __shared__ _Float16 sA2[4][16][HP];

  const int tid = threadIdx.x, w = tid >> 5, lane = tid & 31;

  for (int i = tid; i < HP * 32; i += 128) {
    int n = i >> 5, k = i & 31;
    sW1t[n][k] = (_Float16)((k < DIN && n < HID) ? W1[k * HID + n] : 0.f);
  }
  for (int i = tid; i < HP * HP; i += 128) {
    int n = i / HP, k = i % HP;
    sW2t[n][k] = (_Float16)((k < HID && n < HID) ? W2[k * HID + n] : 0.f);
  }
  for (int i = tid; i < 16 * HP; i += 128) {
    int n = i / HP, k = i % HP;
    sW3t[n][k] = (_Float16)((k < HID && n < EOUT) ? W3[k * EOUT + n] : 0.f);
  }
  for (int i = tid; i < HP; i += 128) {
    sB1[i] = (i < HID) ? B1[i] : 0.f;
    sB2[i] = (i < HID) ? B2[i] : 0.f;
  }
  if (tid < 16) sB3[tid] = (tid < EOUT) ? B3[tid] : 0.f;
  __syncthreads();

  const int ntiles = (N + 15) >> 4;
  const int stride = gridDim.x * 4;
  const int iters  = (ntiles + stride - 1) / stride;

  for (int it = 0; it < iters; ++it) {
    const int tile = (it * gridDim.x + blockIdx.x) * 4 + w;
    if (tile >= ntiles) continue;
    const int base = tile << 4;

    for (int idx = lane; idx < 16 * 32; idx += 32) {
      const int r = idx >> 5, c = idx & 31;
      const int node = base + r;
      float v = 0.f;
      if (node < N) {
        if (c < 7)            v = hin[(size_t)node * 7 + c];
        else if (c < 7 + EIN) v = agg[(size_t)node * EIN + (c - 7)];
      }
      sA0[w][r][c] = (_Float16)v;
    }
    __builtin_amdgcn_wave_barrier();

    v8f acc[NT];
    wmma_gemm<1, NT>(&sA0[w][0][0], 32, &sW1t[0][0], 32, sB1, lane, acc);
    __builtin_amdgcn_wave_barrier();
    store_relu<NT>(&sA1[w][0][0], HP, lane, acc);
    __builtin_amdgcn_wave_barrier();
    wmma_gemm<KT2, NT>(&sA1[w][0][0], HP, &sW2t[0][0], HP, sB2, lane, acc);
    __builtin_amdgcn_wave_barrier();
    store_relu<NT>(&sA2[w][0][0], HP, lane, acc);
    __builtin_amdgcn_wave_barrier();
    v8f o[1];
    wmma_gemm<KT2, 1>(&sA2[w][0][0], HP, &sW3t[0][0], HP, sB3, lane, o);

    const int n  = lane & 15;
    const int mb = (lane & 16) ? 8 : 0;
    if (n < EOUT) {
#pragma unroll
      for (int v = 0; v < 8; ++v) {
        const int node = base + mb + v;
        if (node < N) {
          float x = o[0][v];
          if (RES) x += res[(size_t)node * 7 + n];
          if (SIG) x = 1.f / (1.f + expf(-x));
          out[(size_t)node * EOUT + n] = x;
        }
      }
    }
  }
}

// ---------------------------------------------------------------------------
__global__ void encoder_kernel(const float* __restrict__ x, const float* __restrict__ wq,
                               const float* __restrict__ b, float* __restrict__ h, int N) {
  const int i = blockIdx.x * blockDim.x + threadIdx.x;
  if (i >= N) return;
  float o[7];
#pragma unroll
  for (int j = 0; j < 7; ++j) o[j] = b[j];
#pragma unroll
  for (int k = 0; k < 14; ++k) {
    const float xv = x[(size_t)i * 14 + k];
#pragma unroll
    for (int j = 0; j < 7; ++j) o[j] += xv * wq[k * 7 + j];
  }
#pragma unroll
  for (int j = 0; j < 7; ++j) h[(size_t)i * 7 + j] = o[j];
}

__global__ void zero_kernel(float* __restrict__ p, int n) {
  const int i = blockIdx.x * blockDim.x + threadIdx.x;
  if (i < n) p[i] = 0.f;
}

// ---------------------------------------------------------------------------
extern "C" void kernel_launch(void* const* d_in, const int* in_sizes, int n_in,
                              void* d_out, int out_size, void* d_ws, size_t ws_size,
                              hipStream_t stream) {
  const float* x  = (const float*)d_in[0];
  const int*   ei = (const int*)d_in[1];
  const float* ea = (const float*)d_in[2];
  const int N = in_sizes[0] / 14;
  const int E = in_sizes[2] / 4;
  const int* srcI = ei;
  const int* dstI = ei + E;

  const float* enc_w = (const float*)d_in[3];
  const float* enc_b = (const float*)d_in[4];

  struct P { const float* p[12]; };
  auto blk = [&](int base) { P q; for (int j = 0; j < 12; ++j) q.p[j] = (const float*)d_in[base + j]; return q; };
  P bw1 = blk(5), bw2 = blk(17), bw3 = blk(29);
  P bc1 = blk(41), bc2 = blk(53), bc3 = blk(65);
  const float* Wp[6]; const float* Bp[6]; const float* Xp[6];
  for (int j = 0; j < 6; ++j) {
    Wp[j] = (const float*)d_in[77 + j];
    Bp[j] = (const float*)d_in[83 + j];
    Xp[j] = (const float*)d_in[89 + j];
  }

  float* ws = (float*)d_ws;
  const size_t N7 = (size_t)N * 7, E4 = (size_t)E * 4, E8 = (size_t)E * 8;
  float* h_enc = ws;
  float* h_a   = h_enc + N7;
  float* h_b   = h_a + N7;
  float* e1    = h_b + N7;
  float* e2    = e1 + E4;
  float* e3    = e2 + E4;
  float* ecA   = e3 + E4;
  float* ecB   = ecA + E8;
  float* agg   = ecB + E8;  // N*8 floats max

  float* out_ew   = (float*)d_out;            // [E,1]
  float* out_hc   = out_ew + E;               // [N,8]
  float* out_beta = out_hc + (size_t)N * 8;   // [N,1]

  const int etiles  = (E + 15) / 16;
  const int ntilesN = (N + 15) / 16;
  int eblocks = (etiles + 3) / 4;  if (eblocks > 1200) eblocks = 1200;
  int nblocks = (ntilesN + 3) / 4; if (nblocks > 800)  nblocks = 800;

  auto zero = [&](float* p, int n) {
    zero_kernel<<<(n + 255) / 256, 256, 0, stream>>>(p, n);
  };

  // h = encoder(x)
  encoder_kernel<<<(N + 255) / 256, 256, 0, stream>>>(x, enc_w, enc_b, h_enc, N);

  // ---- in_w1 ----
  zero(agg, N * 4);
  edge_mlp_kernel<14, 4, 64, 64, 4, 0, false, true><<<eblocks, 128, 0, stream>>>(
      h_enc, srcI, dstI, ea, nullptr, nullptr, nullptr, nullptr,
      bw1.p[0], bw1.p[1], bw1.p[2], bw1.p[3], bw1.p[4], bw1.p[5], e1, agg, E);
  node_mlp_kernel<4, 64, 64, 7, false, true><<<nblocks, 128, 0, stream>>>(
      h_enc, agg, bw1.p[6], bw1.p[7], bw1.p[8], bw1.p[9], bw1.p[10], bw1.p[11],
      h_enc, h_a, N);  // h_a = h + h1

  // ---- in_w2 ----
  zero(agg, N * 4);
  edge_mlp_kernel<14, 4, 64, 64, 4, 0, false, true><<<eblocks, 128, 0, stream>>>(
      h_a, srcI, dstI, e1, nullptr, nullptr, nullptr, nullptr,
      bw2.p[0], bw2.p[1], bw2.p[2], bw2.p[3], bw2.p[4], bw2.p[5], e2, agg, E);
  node_mlp_kernel<4, 64, 64, 7, false, true><<<nblocks, 128, 0, stream>>>(
      h_a, agg, bw2.p[6], bw2.p[7], bw2.p[8], bw2.p[9], bw2.p[10], bw2.p[11],
      h_enc, h_b, N);  // h_b = h + h2

  // ---- in_w3 ----
  zero(agg, N * 4);
  edge_mlp_kernel<14, 4, 64, 64, 4, 0, false, true><<<eblocks, 128, 0, stream>>>(
      h_b, srcI, dstI, e2, nullptr, nullptr, nullptr, nullptr,
      bw3.p[0], bw3.p[1], bw3.p[2], bw3.p[3], bw3.p[4], bw3.p[5], e3, agg, E);
  node_mlp_kernel<4, 64, 64, 7, false, true><<<nblocks, 128, 0, stream>>>(
      h_b, agg, bw3.p[6], bw3.p[7], bw3.p[8], bw3.p[9], bw3.p[10], bw3.p[11],
      h_enc, h_a, N);  // h_a = h + h3

  // ---- edge weights: sigmoid(W([edge_attr,e1,e2,e3])), hidden 40 -> pad 64
  edge_mlp_kernel<0, 16, 40, 64, 1, 2, true, false><<<eblocks, 128, 0, stream>>>(
      nullptr, srcI, dstI, ea, e1, e2, e3, nullptr,
      Wp[0], Wp[1], Wp[2], Wp[3], Wp[4], Wp[5], out_ew, nullptr, E);

  // ---- in_c1 (edge feats = [ew, edge_attr, e1, e2, e3] = 17) ----
  zero(agg, N * 8);
  edge_mlp_kernel<14, 17, 64, 64, 8, 1, false, true><<<eblocks, 128, 0, stream>>>(
      h_a, srcI, dstI, ea, e1, e2, e3, out_ew,
      bc1.p[0], bc1.p[1], bc1.p[2], bc1.p[3], bc1.p[4], bc1.p[5], ecA, agg, E);
  node_mlp_kernel<8, 64, 64, 7, false, true><<<nblocks, 128, 0, stream>>>(
      h_a, agg, bc1.p[6], bc1.p[7], bc1.p[8], bc1.p[9], bc1.p[10], bc1.p[11],
      h_enc, h_b, N);  // h_b = h + hc1

  // ---- in_c2 ----
  zero(agg, N * 8);
  edge_mlp_kernel<14, 8, 64, 64, 8, 0, false, true><<<eblocks, 128, 0, stream>>>(
      h_b, srcI, dstI, ecA, nullptr, nullptr, nullptr, nullptr,
      bc2.p[0], bc2.p[1], bc2.p[2], bc2.p[3], bc2.p[4], bc2.p[5], ecB, agg, E);
  node_mlp_kernel<8, 64, 64, 7, false, true><<<nblocks, 128, 0, stream>>>(
      h_b, agg, bc2.p[6], bc2.p[7], bc2.p[8], bc2.p[9], bc2.p[10], bc2.p[11],
      h_enc, h_a, N);  // h_a = h + hc2

  // ---- in_c3 ----
  zero(agg, N * 8);
  edge_mlp_kernel<14, 8, 64, 64, 8, 0, false, true><<<eblocks, 128, 0, stream>>>(
      h_a, srcI, dstI, ecB, nullptr, nullptr, nullptr, nullptr,
      bc3.p[0], bc3.p[1], bc3.p[2], bc3.p[3], bc3.p[4], bc3.p[5], ecA, agg, E);
  node_mlp_kernel<8, 64, 64, 7, false, true><<<nblocks, 128, 0, stream>>>(
      h_a, agg, bc3.p[6], bc3.p[7], bc3.p[8], bc3.p[9], bc3.p[10], bc3.p[11],
      h_enc, h_b, N);  // h_b = hc3 = C3(...) + h

  // ---- beta = sigmoid(B(hc3)), hidden 60 -> pad 64 ----
  node_mlp_kernel<0, 60, 64, 1, true, false><<<nblocks, 128, 0, stream>>>(
      h_b, nullptr, Bp[0], Bp[1], Bp[2], Bp[3], Bp[4], Bp[5],
      nullptr, out_beta, N);

  // ---- hc = X(hc3), hidden 80 -> pad 96 ----
  node_mlp_kernel<0, 80, 96, 8, false, false><<<nblocks, 128, 0, stream>>>(
      h_b, nullptr, Xp[0], Xp[1], Xp[2], Xp[3], Xp[4], Xp[5],
      nullptr, out_hc, N);
}
